// TMix_47485158424858
// MI455X (gfx1250) — matrix operand order repaired and checked by reading, not compiled, not run on hardware
//
#include <hip/hip_runtime.h>
#include <hip/hip_bf16.h>
#include <math.h>

// ---------------------------------------------------------------------------
// RWKV7 time-mix for MI455X (gfx1250, wave32, WMMA + TDM + async-LDS).
// Pipeline:
//   1) mix_kernel: token shift + 6 mixes -> bf16 activation streams
//   2) gemm_bf16_wmma: C(f32) = A(bf16) @ B(bf16); v_wmma_f32_16x16x32_bf16,
//      A tile staged via TDM tensor_load_to_lds (padded rows), B via transpose
//   3) cvt_kernel: f32 -> bf16 (optional tanh / sigmoid) for LoRA mid layers
//   4) assemble_kernel: decay / a / v-mix / kk L2-norm -> scan inputs (B,H,T,N)
//   5) scan_kernel: delta-rule recurrence, S (64x64) in registers per head;
//      per-step vectors staged with global_load_async_to_lds_b32
//   6) out_kernel: GroupNorm + rank-1 bonus + gate -> bf16
//   7) final gemm -> d_out ; v_first copied to d_out second half
// ---------------------------------------------------------------------------

typedef __attribute__((ext_vector_type(16))) __bf16    v16bf;
typedef __attribute__((ext_vector_type(8)))  float     v8f;
typedef __attribute__((ext_vector_type(4)))  unsigned  v4u;
typedef __attribute__((ext_vector_type(8)))  unsigned  v8u;

union Frag32B {
    v16bf v;
    uint4 q[2];
};

#define LOG_W_SCALE (-0.6065306597126334f)
#define GN_EPS      (64e-5f)

__device__ __forceinline__ unsigned short f2bf(float f) {
    unsigned int u = __float_as_uint(f);
    unsigned int r = u + 0x7FFFu + ((u >> 16) & 1u);   // round-to-nearest-even
    return (unsigned short)(r >> 16);
}

__device__ __forceinline__ float sigmoidf_(float x) {
    return 1.0f / (1.0f + __expf(-x));
}

// Generic->LDS byte offset (generic LDS pointers carry the offset in low 32b)
__device__ __forceinline__ unsigned lds_off(const void* p) {
    return (unsigned)(unsigned long long)p;
}

// Async global->LDS copy, 4 bytes per active lane (ASYNCcnt-tracked)
__device__ __forceinline__ void async_ld_b32(unsigned ldsaddr, unsigned long long gaddr) {
    asm volatile("global_load_async_to_lds_b32 %0, %1, off"
                 :: "v"(ldsaddr), "v"(gaddr) : "memory");
}
__device__ __forceinline__ void wait_asynccnt0() {
    asm volatile("s_wait_asynccnt 0x0" ::: "memory");
}

// ---------------------------------------------------------------------------
// 1) token shift + 6-way mix, f32 -> bf16
// ---------------------------------------------------------------------------
__global__ __launch_bounds__(256) void mix_kernel(
    const float* __restrict__ x, const float* __restrict__ x_mix,
    unsigned short* __restrict__ dst,   // 6 planes of MD bf16
    int T, int D, long long MD)
{
    long long i = (long long)blockIdx.x * 256 + threadIdx.x;
    if (i >= MD) return;
    int d = (int)(i % D);
    long long m = i / D;
    int t = (int)(m % T);
    float xv   = x[i];
    float prev = (t > 0) ? x[i - D] : 0.0f;
    float delta = prev - xv;
    #pragma unroll
    for (int j = 0; j < 6; ++j)
        dst[(long long)j * MD + i] = f2bf(xv + delta * x_mix[j * D + d]);
}

// ---------------------------------------------------------------------------
// f32 -> bf16 convert with optional activation (0: none, 1: tanh, 2: sigmoid)
// ---------------------------------------------------------------------------
__global__ __launch_bounds__(256) void cvt_kernel(
    const float* __restrict__ src, unsigned short* __restrict__ dst,
    long long n, int act)
{
    long long i = (long long)blockIdx.x * 256 + threadIdx.x;
    if (i >= n) return;
    float v = src[i];
    if (act == 1)      v = tanhf(v);
    else if (act == 2) v = sigmoidf_(v);
    dst[i] = f2bf(v);
}

// ---------------------------------------------------------------------------
// 2) Tiled bf16 WMMA GEMM:  C[M,N](f32) = A[M,K](bf16,row) @ B[K,N](bf16,row)
//    Tile: 128(M) x 64(N), K-chunks of 32. 256 threads = 8 waves; wave w owns
//    rows [16w,16w+16) x all 64 cols = 4 v_wmma_f32_16x16x32_bf16 per chunk.
//
//    A tile staged by the Tensor Data Mover: one tensor_load_to_lds per chunk
//    copies the 128x32-half tile; TDM LDS padding (16 DWORDs data -> +8 DWORD
//    pad) produces the 48-half padded row stride, so fragment reads remain
//    16B-aligned ds_load_b128 matching the ISA 16-bit A layout:
//      lane<16: M=lane, K in {0..7, 16..23}  (+8 for lanes 16..31)
//    B staged transposed in LDS: lane<16: N=lane, K in {0..7, 16..23} (+8 for lanes 16..31).
//    Requires M % 128 == 0, K % 32 == 0 (true for all call sites).
// ---------------------------------------------------------------------------
#define GBM 128
#define GBN 64
#define GBK 32
#define APAD 48   // halves per A row in LDS (96B: 64B data + 32B TDM pad)
#define BPAD 48   // halves per B col in LDS

__global__ __launch_bounds__(256) void gemm_bf16_wmma(
    const unsigned short* __restrict__ A,
    const unsigned short* __restrict__ B,
    float* __restrict__ C,
    int M, int N, int K)
{
    __shared__ unsigned short As[GBM][APAD];  // [m][k]  (TDM-filled)
    __shared__ unsigned short Bs[GBN][BPAD];  // [n][k]  (transposed)

    const int tid  = threadIdx.x;
    const int wave = tid >> 5;
    const int lane = tid & 31;
    const int half = lane >> 4;
    const int l15  = lane & 15;
    const int m0 = blockIdx.y * GBM;
    const int n0 = blockIdx.x * GBN;

    v8f acc[4];
    #pragma unroll
    for (int i = 0; i < 4; ++i)
        #pragma unroll
        for (int j = 0; j < 8; ++j) acc[i][j] = 0.0f;

    const unsigned as_base = lds_off(&As[0][0]);

    for (int k0 = 0; k0 < K; k0 += GBK) {
        // software prefetch of the next K-chunk (global_prefetch_b8)
        if (k0 + GBK < K) {
            __builtin_prefetch(A + (size_t)(m0 + (tid >> 1)) * K + k0 + GBK, 0, 1);
            __builtin_prefetch(B + (size_t)(k0 + GBK + (tid & 31)) * N + n0, 0, 1);
        }

        // ---- A tile via TDM: 2D tile 32(x) x 128(y) halves, row stride K.
        if (wave == 0) {
            unsigned long long ga =
                (unsigned long long)(A + (size_t)m0 * K + k0);
            v4u g0;
            g0.x = 1u;                                  // count=1, user mode
            g0.y = as_base;                             // lds_addr
            g0.z = (unsigned)(ga & 0xFFFFFFFFull);      // global_addr[31:0]
            g0.w = (unsigned)((ga >> 32) & 0x1FFFFFFull)// global_addr[56:32]
                 | 0x80000000u;                         // type=2 ("image")
            v8u g1;
            g1[0] = (1u << 16)      // data_size = 2 bytes
                  | (1u << 20)      // pad_enable
                  | (3u << 22)      // pad_interval: 16 DWORDs between pads
                  | (7u << 25);     // pad_amount:   8 DWORDs of pad
            g1[1] = ((unsigned)K & 0xFFFFu) << 16;          // tensor_dim0 lo16
            g1[2] = (((unsigned)K >> 16) & 0xFFFFu)         // tensor_dim0 hi16
                  | ((unsigned)GBM << 16);                  // tensor_dim1 lo16
            g1[3] = ((unsigned)GBK << 16);                  // tile_dim0 = 32
            g1[4] = (unsigned)GBM;                          // tile_dim1 = 128
            g1[5] = (unsigned)K;                            // dim0_stride lo32
            g1[6] = 0u;                                     // stride hi / d1s
            g1[7] = 0u;
            asm volatile("tensor_load_to_lds %0, %1"
                         :: "s"(g0), "s"(g1) : "memory");
            __builtin_amdgcn_s_wait_tensorcnt(0);
        }

        // ---- B tile: 32 x 64, one 16B vector per thread, transposed store
        {
            int row = tid >> 3;          // k within chunk
            int col = (tid & 7) * 8;     // n within tile
            uint4 dv;
            if (n0 + col + 8 <= N) {
                dv = *(const uint4*)(B + (size_t)(k0 + row) * N + n0 + col);
            } else {
                dv.x = dv.y = dv.z = dv.w = 0u;
            }
            const unsigned short* hh = (const unsigned short*)&dv;
            #pragma unroll
            for (int j = 0; j < 8; ++j) Bs[col + j][row] = hh[j];
        }
        __syncthreads();

        // ---- A fragment for this wave's 16-row strip
        Frag32B af;
        const int am = wave * 16 + l15;
        af.q[0] = *(const uint4*)&As[am][half * 8];        // K 0..7   (+8)
        af.q[1] = *(const uint4*)&As[am][16 + half * 8];   // K 16..23 (+8)

        // ---- 4 N-tiles of 16
        #pragma unroll
        for (int nt = 0; nt < 4; ++nt) {
            Frag32B bf;
            const int bn = nt * 16 + l15;
            bf.q[0] = *(const uint4*)&Bs[bn][half * 8];        // K 0..7   (+8)
            bf.q[1] = *(const uint4*)&Bs[bn][16 + half * 8];   // K 16..23 (+8)
            acc[nt] = __builtin_amdgcn_wmma_f32_16x16x32_bf16(
                false, af.v, false, bf.v, (short)0, acc[nt], false, false);
        }
        __syncthreads();
    }

    // ---- store: C/D layout lane<16 -> M=i, lane>=16 -> M=i+8, N=l15
    #pragma unroll
    for (int nt = 0; nt < 4; ++nt) {
        int col = n0 + nt * 16 + l15;
        if (col < N) {
            #pragma unroll
            for (int i = 0; i < 8; ++i) {
                int row = m0 + wave * 16 + half * 8 + i;
                C[(size_t)row * N + col] = acc[nt][i];
            }
        }
    }
}

// ---------------------------------------------------------------------------
// 4) Per-(b,t,h) assembly: decay, a, value-mix, kk L2-norm; write scan inputs
//    in (B,H,T,N) layout. One 64-thread block per (b,t,h).
// ---------------------------------------------------------------------------
__global__ __launch_bounds__(64) void assemble_kernel(
    const float* __restrict__ r_in, const float* __restrict__ k_in,
    const float* __restrict__ v_in, const float* __restrict__ lw_lin,
    const float* __restrict__ a_lin, const float* __restrict__ v_lin,
    const float* __restrict__ v_first,
    const float* __restrict__ w0, const float* __restrict__ a0,
    const float* __restrict__ v0,
    const float* __restrict__ k_k, const float* __restrict__ k_a,
    float* __restrict__ r_s, float* __restrict__ w_s, float* __restrict__ k_s,
    float* __restrict__ v_s, float* __restrict__ a_s, float* __restrict__ b_s,
    int T, int H, int N)
{
    __shared__ float red[64];
    const int n = threadIdx.x;
    const int h = blockIdx.x % H;
    const long long m = blockIdx.x / H;       // b*T + t
    const int t = (int)(m % T);
    const long long b = m / T;
    const int d = h * N + n;
    const long long idx = m * (long long)(H * N) + d;

    float rf = r_in[idx];
    float kf = k_in[idx];
    float vr = v_in[idx];

    float dec  = __expf(LOG_W_SCALE * sigmoidf_(w0[d] + lw_lin[idx]));
    float a    = sigmoidf_(a0[d] + a_lin[idx]);
    float vmix = sigmoidf_(v0[d] + v_lin[idx]);
    float v    = vr + (v_first[idx] - vr) * vmix;

    float kk = kf * k_k[d];
    red[n] = kk * kk;
    __syncthreads();
    float s = 0.0f;
    for (int j = 0; j < 64; ++j) s += red[j];
    float inv = 1.0f / fmaxf(sqrtf(s), 1e-12f);
    float kkn = kk * inv;
    float kfin = kf * (1.0f + (a - 1.0f) * k_a[d]);

    const long long sidx = (((b * H + h) * T) + t) * (long long)N + n;
    r_s[sidx] = rf;
    w_s[sidx] = dec;
    k_s[sidx] = kfin;
    v_s[sidx] = v;
    a_s[sidx] = -kkn;
    b_s[sidx] = kkn * a;
}

// ---------------------------------------------------------------------------
// 5) Delta-rule scan. One block per (b,h); thread j owns state row S[j][0..63]
//    in registers.  S = S*diag(w) + (S a) b^T + v k^T ;  o = S r
//    Per-step vectors staged global->LDS with async copies (ASYNCcnt).
// ---------------------------------------------------------------------------
__global__ __launch_bounds__(64) void scan_kernel(
    const float* __restrict__ r_s, const float* __restrict__ w_s,
    const float* __restrict__ k_s, const float* __restrict__ v_s,
    const float* __restrict__ a_s, const float* __restrict__ b_s,
    float* __restrict__ o_s, int T, int N)
{
    __shared__ float rv[64], wv[64], kv[64], vv[64], av[64], bv[64];
    const int j = threadIdx.x;
    const long long base = (long long)blockIdx.x * T * N;

    float S[64];
    #pragma unroll
    for (int k = 0; k < 64; ++k) S[k] = 0.0f;

    for (int t = 0; t < T; ++t) {
        const long long off = base + (long long)t * N + j;
        async_ld_b32(lds_off(&rv[j]), (unsigned long long)(r_s + off));
        async_ld_b32(lds_off(&wv[j]), (unsigned long long)(w_s + off));
        async_ld_b32(lds_off(&kv[j]), (unsigned long long)(k_s + off));
        async_ld_b32(lds_off(&vv[j]), (unsigned long long)(v_s + off));
        async_ld_b32(lds_off(&av[j]), (unsigned long long)(a_s + off));
        async_ld_b32(lds_off(&bv[j]), (unsigned long long)(b_s + off));
        wait_asynccnt0();
        __syncthreads();

        float sa = 0.0f;
        #pragma unroll
        for (int k = 0; k < 64; ++k) sa += S[k] * av[k];

        const float vj = vv[j];
        float o = 0.0f;
        #pragma unroll
        for (int k = 0; k < 64; ++k) {
            float s = S[k] * wv[k] + sa * bv[k] + vj * kv[k];
            S[k] = s;
            o += s * rv[k];
        }
        o_s[off] = o;
        __syncthreads();
    }
}

// ---------------------------------------------------------------------------
// 6) GroupNorm + rank-1 bonus + gate -> bf16 (input to final GEMM)
// ---------------------------------------------------------------------------
__global__ __launch_bounds__(64) void out_kernel(
    const float* __restrict__ o_s, const float* __restrict__ r_s,
    const float* __restrict__ k_s, const float* __restrict__ v_s,
    const float* __restrict__ g_in, const float* __restrict__ r_k,
    const float* __restrict__ gn_w, const float* __restrict__ gn_b,
    unsigned short* __restrict__ og, int T, int H, int N)
{
    __shared__ float red[64], red2[64];
    const int n = threadIdx.x;
    const int h = blockIdx.x % H;
    const long long m = blockIdx.x / H;
    const int t = (int)(m % T);
    const long long b = m / T;
    const long long sidx = (((b * H + h) * T) + t) * (long long)N + n;

    float ov = o_s[sidx];
    red[n] = ov; red2[n] = ov * ov;
    __syncthreads();
    float s1 = 0.0f, s2 = 0.0f;
    for (int q = 0; q < 64; ++q) { s1 += red[q]; s2 += red2[q]; }
    float mu  = s1 * (1.0f / 64.0f);
    float var = s2 * (1.0f / 64.0f) - mu * mu;
    float oo = (ov - mu) * rsqrtf(var + GN_EPS);
    const int d = h * N + n;
    oo = oo * gn_w[d] + gn_b[d];
    __syncthreads();

    float rf = r_s[sidx], kf = k_s[sidx], vf = v_s[sidx];
    red[n] = rf * kf * r_k[h * N + n];
    __syncthreads();
    float dot = 0.0f;
    for (int q = 0; q < 64; ++q) dot += red[q];

    const long long idx = m * (long long)(H * N) + d;
    float val = (oo + dot * vf) * g_in[idx];
    og[idx] = f2bf(val);
}

// ---------------------------------------------------------------------------
// Host orchestration
// ---------------------------------------------------------------------------
extern "C" void kernel_launch(void* const* d_in, const int* in_sizes, int n_in,
                              void* d_out, int out_size, void* d_ws, size_t ws_size,
                              hipStream_t stream)
{
    const int B = 8, T = 1024, D = 1024, H = 16, N = 64;
    const int D_W = 64, D_A = 64, D_V = 32, D_G = 128;
    const long long M  = (long long)B * T;     // 8192
    const long long MD = M * D;                // 8.39M

    const float* x     = (const float*)d_in[0];
    const float* vfst  = (const float*)d_in[1];
    const float* xmix  = (const float*)d_in[2];
    const float* Wr    = (const float*)d_in[3];
    const float* Wk    = (const float*)d_in[4];
    const float* Wv    = (const float*)d_in[5];
    const float* Wo    = (const float*)d_in[6];
    const float* w0    = (const float*)d_in[7];
    const float* w1    = (const float*)d_in[8];
    const float* w2    = (const float*)d_in[9];
    const float* a0    = (const float*)d_in[10];
    const float* a1    = (const float*)d_in[11];
    const float* a2    = (const float*)d_in[12];
    const float* v0    = (const float*)d_in[13];
    const float* v1    = (const float*)d_in[14];
    const float* v2    = (const float*)d_in[15];
    const float* g1    = (const float*)d_in[16];
    const float* g2    = (const float*)d_in[17];
    const float* k_k   = (const float*)d_in[18];
    const float* k_a   = (const float*)d_in[19];
    const float* r_k   = (const float*)d_in[20];
    const float* gn_w  = (const float*)d_in[21];
    const float* gn_b  = (const float*)d_in[22];

    // ---- workspace carving (256B aligned)
    char* wsb = (char*)d_ws;
    size_t off = 0;
    auto alloc = [&](size_t bytes) -> char* {
        off = (off + 255) & ~(size_t)255;
        char* p = wsb + off;
        off += bytes;
        return p;
    };
    unsigned short* mix6 = (unsigned short*)alloc(6 * MD * 2);   // xr,xw,xk,xv,xa,xg
    unsigned short* wrb  = (unsigned short*)alloc((size_t)D * D * 2);
    unsigned short* wkb  = (unsigned short*)alloc((size_t)D * D * 2);
    unsigned short* wvb  = (unsigned short*)alloc((size_t)D * D * 2);
    unsigned short* wob  = (unsigned short*)alloc((size_t)D * D * 2);
    unsigned short* w1b  = (unsigned short*)alloc((size_t)D * D_W * 2);
    unsigned short* w2b  = (unsigned short*)alloc((size_t)D_W * D * 2);
    unsigned short* a1b  = (unsigned short*)alloc((size_t)D * D_A * 2);
    unsigned short* a2b  = (unsigned short*)alloc((size_t)D_A * D * 2);
    unsigned short* v1b  = (unsigned short*)alloc((size_t)D * D_V * 2);
    unsigned short* v2b  = (unsigned short*)alloc((size_t)D_V * D * 2);
    unsigned short* g1b  = (unsigned short*)alloc((size_t)D * D_G * 2);
    unsigned short* g2b  = (unsigned short*)alloc((size_t)D_G * D * 2);

    float* r_f  = (float*)alloc(MD * 4);
    float* k_f  = (float*)alloc(MD * 4);
    float* vr_f = (float*)alloc(MD * 4);
    float* hw_f = (float*)alloc(M * D_W * 4);
    float* ha_f = (float*)alloc(M * D_A * 4);
    float* hv_f = (float*)alloc(M * D_V * 4);
    float* hg_f = (float*)alloc(M * D_G * 4);
    unsigned short* hw_b = (unsigned short*)alloc(M * D_W * 2);
    unsigned short* ha_b = (unsigned short*)alloc(M * D_A * 2);
    unsigned short* hv_b = (unsigned short*)alloc(M * D_V * 2);
    unsigned short* hg_b = (unsigned short*)alloc(M * D_G * 2);
    float* lw_f = (float*)alloc(MD * 4);
    float* al_f = (float*)alloc(MD * 4);
    float* vl_f = (float*)alloc(MD * 4);
    float* g_f  = (float*)alloc(MD * 4);

    float* r_s  = (float*)alloc(MD * 4);     // (B,H,T,N) layouts
    float* w_s  = (float*)alloc(MD * 4);
    float* k_s  = (float*)alloc(MD * 4);
    float* v_s  = (float*)alloc(MD * 4);
    float* a_s  = (float*)alloc(MD * 4);
    float* b_s  = (float*)alloc(MD * 4);
    float* o_s  = (float*)alloc(MD * 4);
    unsigned short* og_b = (unsigned short*)alloc(MD * 2);
    (void)ws_size; (void)in_sizes; (void)n_in; (void)out_size;

    auto cvt = [&](const float* src, unsigned short* dst, long long n, int act) {
        cvt_kernel<<<dim3((unsigned)((n + 255) / 256)), 256, 0, stream>>>(src, dst, n, act);
    };
    auto gemm = [&](const unsigned short* A, const unsigned short* Bm, float* C,
                    int m, int n, int k) {
        dim3 grid((unsigned)((n + GBN - 1) / GBN), (unsigned)(m / GBM));
        gemm_bf16_wmma<<<grid, 256, 0, stream>>>(A, Bm, C, m, n, k);
    };

    // ---- weight conversions to bf16
    cvt(Wr, wrb, (long long)D * D, 0);
    cvt(Wk, wkb, (long long)D * D, 0);
    cvt(Wv, wvb, (long long)D * D, 0);
    cvt(Wo, wob, (long long)D * D, 0);
    cvt(w1, w1b, (long long)D * D_W, 0);
    cvt(w2, w2b, (long long)D_W * D, 0);
    cvt(a1, a1b, (long long)D * D_A, 0);
    cvt(a2, a2b, (long long)D_A * D, 0);
    cvt(v1, v1b, (long long)D * D_V, 0);
    cvt(v2, v2b, (long long)D_V * D, 0);
    cvt(g1, g1b, (long long)D * D_G, 0);
    cvt(g2, g2b, (long long)D_G * D, 0);

    // ---- token shift + mix
    mix_kernel<<<dim3((unsigned)((MD + 255) / 256)), 256, 0, stream>>>(
        x, xmix, mix6, T, D, MD);

    const unsigned short* xr = mix6 + 0 * MD;
    const unsigned short* xw = mix6 + 1 * MD;
    const unsigned short* xk = mix6 + 2 * MD;
    const unsigned short* xv = mix6 + 3 * MD;
    const unsigned short* xa = mix6 + 4 * MD;
    const unsigned short* xg = mix6 + 5 * MD;

    // ---- projections
    gemm(xr, wrb, r_f,  (int)M, D, D);
    gemm(xk, wkb, k_f,  (int)M, D, D);
    gemm(xv, wvb, vr_f, (int)M, D, D);

    // ---- LoRA chains
    gemm(xw, w1b, hw_f, (int)M, D_W, D);
    cvt(hw_f, hw_b, M * D_W, 1);                 // tanh
    gemm(hw_b, w2b, lw_f, (int)M, D, D_W);

    gemm(xa, a1b, ha_f, (int)M, D_A, D);
    cvt(ha_f, ha_b, M * D_A, 0);
    gemm(ha_b, a2b, al_f, (int)M, D, D_A);

    gemm(xv, v1b, hv_f, (int)M, D_V, D);
    cvt(hv_f, hv_b, M * D_V, 0);
    gemm(hv_b, v2b, vl_f, (int)M, D, D_V);

    gemm(xg, g1b, hg_f, (int)M, D_G, D);
    cvt(hg_f, hg_b, M * D_G, 2);                 // sigmoid
    gemm(hg_b, g2b, g_f, (int)M, D, D_G);

    // ---- per-head assembly into scan layout
    assemble_kernel<<<dim3((unsigned)(M * H)), 64, 0, stream>>>(
        r_f, k_f, vr_f, lw_f, al_f, vl_f, vfst,
        w0, a0, v0, k_k, k_a,
        r_s, w_s, k_s, v_s, a_s, b_s, T, H, N);

    // ---- sequential delta-rule scan (128 heads in parallel)
    scan_kernel<<<dim3((unsigned)(B * H)), 64, 0, stream>>>(
        r_s, w_s, k_s, v_s, a_s, b_s, o_s, T, N);

    // ---- GroupNorm + bonus + gate
    out_kernel<<<dim3((unsigned)(M * H)), 64, 0, stream>>>(
        o_s, r_s, k_s, v_s, g_f, r_k, gn_w, gn_b, og_b, T, H, N);

    // ---- final projection into d_out (first tuple element)
    gemm(og_b, wob, (float*)d_out, (int)M, D, D);

    // ---- second tuple element: v_first passthrough
    hipMemcpyAsync((float*)d_out + MD, (const void*)vfst, MD * sizeof(float),
                   hipMemcpyDeviceToDevice, stream);
}
